// SelfAttention_77790447665435
// MI455X (gfx1250) — matrix operand order repaired and checked by reading, not compiled, not run on hardware
//
#include <hip/hip_runtime.h>
#include <hip/hip_bf16.h>

// ---------------------------------------------------------------------------
// Self-attention (B=2, S=4096, D=256, H=4, DH=64) for gfx1250 (MI455X).
// bf16 WMMA everywhere; K/V tiles staged to LDS with async-to-LDS (ASYNCcnt)
// double buffering, shared by 4 waves per workgroup.
// ---------------------------------------------------------------------------

typedef __bf16 bf16_t;
typedef __attribute__((ext_vector_type(16))) __bf16 v16bf;
typedef __attribute__((ext_vector_type(8)))  __bf16 v8bf;
typedef __attribute__((ext_vector_type(8)))  float  v8f;
typedef __attribute__((ext_vector_type(4)))  int    v4i;

#define WMMA_BF16(A, B, C) \
  __builtin_amdgcn_wmma_f32_16x16x32_bf16(false, (A), false, (B), (short)0, (C), false, false)

constexpr int Bb = 2, S = 4096, D = 256, H = 4, DH = 64;
constexpr int M_TOT = Bb * S;              // 8192 token rows

#if defined(__gfx1250__) && __has_builtin(__builtin_amdgcn_global_load_async_to_lds_b128)
#define USE_ASYNC_LDS 1
typedef __attribute__((address_space(1))) v4i* gas1_t;   // global int4*
typedef __attribute__((address_space(3))) v4i* las3_t;   // LDS int4*
#else
#define USE_ASYNC_LDS 0
#endif

// ---------------------------------------------------------------------------
// Kernel 1: QKV projection.  Y = X * W^T + b  (W stored [out,in] per torch).
// One wave computes one 16x16 tile of one of {Q,K,V}.
// Q,K written bf16 as [B,H,S,DH]; V written bf16 TRANSPOSED as [B,H,DH,S].
// ---------------------------------------------------------------------------
__global__ __launch_bounds__(32)
void qkv_proj_kernel(const float* __restrict__ x,
                     const float* __restrict__ Wq, const float* __restrict__ bq,
                     const float* __restrict__ Wk, const float* __restrict__ bk,
                     const float* __restrict__ Wv, const float* __restrict__ bv,
                     bf16_t* __restrict__ Qw, bf16_t* __restrict__ Kw,
                     bf16_t* __restrict__ Vtw)
{
    const int lane = threadIdx.x;
    const int g    = lane >> 4;        // half-wave group
    const int ln   = lane & 15;
    const int mtile = blockIdx.x;      // 0..511
    const int etile = blockIdx.y & 15; // 0..15
    const int mat   = blockIdx.y >> 4; // 0=Q 1=K 2=V

    const float* W;  const float* bias;
    if (mat == 0)      { W = Wq; bias = bq; }
    else if (mat == 1) { W = Wk; bias = bk; }
    else               { W = Wv; bias = bv; }

    const int m = mtile * 16 + ln;     // A-row (token), same for both halves
    const int e = etile * 16 + ln;     // B-col = W row = output feature
    const float* xrow = x + (size_t)m * D;
    const float* wrow = W + (size_t)e * D;

    v8f acc = {};
    #pragma unroll
    for (int kc = 0; kc < D / 32; ++kc) {
        // A: X[m, kc*32 + 16g + 0..15] (16 contiguous)
        v16bf a;
        const float* pa0 = xrow + kc * 32 + 16 * g;
        const float* pa1 = xrow + kc * 32 + 16 * g + 8;
        #pragma unroll
        for (int i = 0; i < 8; ++i) { a[i] = (bf16_t)pa0[i]; a[8 + i] = (bf16_t)pa1[i]; }
        // B: W[e, kc*32 + 16g + 0..15] (16 contiguous)
        v16bf bm;
        const float* pb = wrow + kc * 32 + 16 * g;
        #pragma unroll
        for (int i = 0; i < 16; ++i) bm[i] = (bf16_t)pb[i];
        acc = WMMA_BF16(a, bm, acc);
    }

    const float bval = bias[e];
    const int h = e >> 6, dh = e & (DH - 1);

    if (mat < 2) {
        bf16_t* dst = (mat == 0) ? Qw : Kw;
        #pragma unroll
        for (int r = 0; r < 8; ++r) {
            const int mm = mtile * 16 + 8 * g + r;       // C row M = r + 8g
            const int b = mm >> 12, s = mm & (S - 1);
            dst[(((size_t)(b * H + h) * S + s) * DH) + dh] = (bf16_t)(acc[r] + bval);
        }
    } else {
        const int mb = mtile * 16 + 8 * g;
        const int b = mb >> 12, s0 = mb & (S - 1);
        v8bf vv;
        #pragma unroll
        for (int r = 0; r < 8; ++r) vv[r] = (bf16_t)(acc[r] + bval);
        *reinterpret_cast<v8bf*>(Vtw + ((size_t)(b * H + h) * DH + dh) * S + s0) = vv;
    }
}

// ---------------------------------------------------------------------------
// Stage one 32-key chunk of K ([32,64] bf16, 4KB, contiguous in global) and
// V^T ([64,32] bf16, 4KB, 64 rows of 64B) into LDS. 128 threads, 16B/thread
// per issue, 4 issues each.  Async path uses GLOBAL_LOAD_ASYNC_TO_LDS_B128.
// ---------------------------------------------------------------------------
__device__ __forceinline__
void stage_chunk(const bf16_t* __restrict__ Kb, const bf16_t* __restrict__ Vt,
                 int kb, bf16_t* kdst, bf16_t* vdst, int t)
{
    #pragma unroll
    for (int j = 0; j < 2; ++j) {
        const int idx = j * 128 + t;                       // 0..255
        const bf16_t* gk = Kb + (size_t)kb * DH + idx * 8; // contiguous 4KB
        bf16_t*       lk = kdst + idx * 8;
        const int row = idx >> 2, c = (idx & 3) * 8;       // V^T row / col
        const bf16_t* gv = Vt + (size_t)row * S + kb + c;
        bf16_t*       lv = vdst + row * 32 + c;
#if USE_ASYNC_LDS
        __builtin_amdgcn_global_load_async_to_lds_b128(
            (gas1_t)(uintptr_t)gk, (las3_t)(uint32_t)(uintptr_t)lk, 0, 0);
        __builtin_amdgcn_global_load_async_to_lds_b128(
            (gas1_t)(uintptr_t)gv, (las3_t)(uint32_t)(uintptr_t)lv, 0, 0);
#else
        *reinterpret_cast<v8bf*>(lk) = *reinterpret_cast<const v8bf*>(gk);
        *reinterpret_cast<v8bf*>(lv) = *reinterpret_cast<const v8bf*>(gv);
#endif
    }
}

__device__ __forceinline__ void wait_stage()
{
#if USE_ASYNC_LDS
    __builtin_amdgcn_s_wait_asynccnt(0);
#endif
    __syncthreads();
}

// ---------------------------------------------------------------------------
// Kernel 2: flash attention. 4 waves per workgroup share one head's K/V
// chunks via LDS; each wave owns one 16-query tile.  Computes S^T = K·Q^T
// (lane = query column => per-lane scalar softmax state); P^T B-operand for
// out^T = V^T·P^T rebuilt with shfl_xor(16) + in-lane bf16 packs.
// ---------------------------------------------------------------------------
__global__ __launch_bounds__(128)
void flash_attn_kernel(const bf16_t* __restrict__ Qw, const bf16_t* __restrict__ Kw,
                       const bf16_t* __restrict__ Vtw, float* __restrict__ out)
{
    __shared__ bf16_t kbuf[2][32 * DH];    // [key][dh]
    __shared__ bf16_t vbuf[2][DH * 32];    // [dh][key]

    const int t = threadIdx.x;
    const int wave = t >> 5, lane = t & 31;
    const int g = lane >> 4, ln = lane & 15;
    const int qtile = blockIdx.x * 4 + wave;   // 0..255
    const int bh    = blockIdx.y;              // b*H + h
    const int b = bh >> 2, h = bh & (H - 1);

    const bf16_t* Qb = Qw  + (size_t)bh * S * DH;
    const bf16_t* Kb = Kw  + (size_t)bh * S * DH;
    const bf16_t* Vt = Vtw + (size_t)bh * DH * S;

    // Q as B operand (lane col = query), two 32-wide dh chunks
    const bf16_t* qrow = Qb + (size_t)(qtile * 16 + ln) * DH;
    v16bf qb0, qb1;
    {
        v8bf q00 = *reinterpret_cast<const v8bf*>(qrow + 8 * g);
        v8bf q01 = *reinterpret_cast<const v8bf*>(qrow + 16 + 8 * g);
        v8bf q10 = *reinterpret_cast<const v8bf*>(qrow + 32 + 8 * g);
        v8bf q11 = *reinterpret_cast<const v8bf*>(qrow + 48 + 8 * g);
        #pragma unroll
        for (int i = 0; i < 8; ++i) {
            qb0[i] = q00[i]; qb0[8 + i] = q01[i];
            qb1[i] = q10[i]; qb1[8 + i] = q11[i];
        }
    }

    float mrun = -3.0e38f, lsum = 0.0f;
    v8f accv[4] = {v8f{}, v8f{}, v8f{}, v8f{}};
    const float scale = 0.125f;            // 1/sqrt(DH)

    stage_chunk(Kb, Vt, 0, kbuf[0], vbuf[0], t);
    wait_stage();

    int p = 0;
    for (int it = 0; it < S / 32; ++it) {
        const int kb = it * 32;
        if (it + 1 < S / 32)
            stage_chunk(Kb, Vt, kb + 32, kbuf[p ^ 1], vbuf[p ^ 1], t);

        // ---- scores^T: two 16-key tiles from LDS ----
        v8f s0 = {}, s1 = {};
        {
            const bf16_t* kr0 = &kbuf[p][ln * DH];
            const bf16_t* kr1 = &kbuf[p][(16 + ln) * DH];
            v16bf a;
            #pragma unroll
            for (int kc = 0; kc < 2; ++kc) {
                v8bf lo = *reinterpret_cast<const v8bf*>(kr0 + kc * 32 + 8 * g);
                v8bf hi = *reinterpret_cast<const v8bf*>(kr0 + kc * 32 + 16 + 8 * g);
                #pragma unroll
                for (int i = 0; i < 8; ++i) { a[i] = lo[i]; a[8 + i] = hi[i]; }
                s0 = WMMA_BF16(a, (kc == 0 ? qb0 : qb1), s0);
            }
            #pragma unroll
            for (int kc = 0; kc < 2; ++kc) {
                v8bf lo = *reinterpret_cast<const v8bf*>(kr1 + kc * 32 + 8 * g);
                v8bf hi = *reinterpret_cast<const v8bf*>(kr1 + kc * 32 + 16 + 8 * g);
                #pragma unroll
                for (int i = 0; i < 8; ++i) { a[i] = lo[i]; a[8 + i] = hi[i]; }
                s1 = WMMA_BF16(a, (kc == 0 ? qb0 : qb1), s1);
            }
        }

        // ---- online softmax (per-lane scalar state; lane = one query) ----
        float cmax = s0[0];
        #pragma unroll
        for (int r = 1; r < 8; ++r) cmax = fmaxf(cmax, s0[r]);
        #pragma unroll
        for (int r = 0; r < 8; ++r) cmax = fmaxf(cmax, s1[r]);
        cmax = fmaxf(cmax, __shfl_xor(cmax, 16, 32));
        const float mnew  = fmaxf(mrun, cmax);
        const float alpha = __expf(scale * (mrun - mnew));

        float p0[8], p1[8], psum = 0.0f;
        #pragma unroll
        for (int r = 0; r < 8; ++r) {
            p0[r] = __expf(scale * (s0[r] - mnew));
            p1[r] = __expf(scale * (s1[r] - mnew));
            psum += p0[r] + p1[r];
        }
        psum += __shfl_xor(psum, 16, 32);
        lsum = lsum * alpha + psum;
        mrun = mnew;

        // ---- build P^T B-operand (lane needs local keys 16g..16g+15) ----
        v16bf pt;
        #pragma unroll
        for (int r = 0; r < 8; ++r) {
            const float o0 = __shfl_xor(p0[r], 16, 32);
            const float o1 = __shfl_xor(p1[r], 16, 32);
            pt[r]     = (bf16_t)((g == 0) ? p0[r] : o1);    // keys 16g + 0..7
            pt[8 + r] = (bf16_t)((g == 0) ? o0    : p1[r]); // keys 16g + 8..15
        }

        // ---- out^T += V^T · P^T from LDS (rescale accumulators) ----
        #pragma unroll
        for (int dt = 0; dt < 4; ++dt) {
            #pragma unroll
            for (int r = 0; r < 8; ++r) accv[dt][r] *= alpha;
            const bf16_t* vrow = &vbuf[p][(dt * 16 + ln) * 32];
            v8bf lo = *reinterpret_cast<const v8bf*>(vrow + 16 * g);
            v8bf hi = *reinterpret_cast<const v8bf*>(vrow + 16 * g + 8);
            v16bf a;
            #pragma unroll
            for (int i = 0; i < 8; ++i) { a[i] = lo[i]; a[8 + i] = hi[i]; }
            accv[dt] = WMMA_BF16(a, pt, accv[dt]);
        }

        wait_stage();         // chunk i+1 visible; all reads of buf[p] done
        p ^= 1;
    }

    // ---- normalize and store: lane = query s, rows = contiguous dh ----
    const float inv = 1.0f / lsum;
    const int srow = qtile * 16 + ln;
    float* orow = out + ((size_t)b * S + srow) * D + h * DH;
    #pragma unroll
    for (int dt = 0; dt < 4; ++dt) {
        float4 v0, v1;
        v0.x = accv[dt][0] * inv; v0.y = accv[dt][1] * inv;
        v0.z = accv[dt][2] * inv; v0.w = accv[dt][3] * inv;
        v1.x = accv[dt][4] * inv; v1.y = accv[dt][5] * inv;
        v1.z = accv[dt][6] * inv; v1.w = accv[dt][7] * inv;
        *reinterpret_cast<float4*>(orow + dt * 16 + 8 * g)     = v0;
        *reinterpret_cast<float4*>(orow + dt * 16 + 8 * g + 4) = v1;
    }
}

// ---------------------------------------------------------------------------
extern "C" void kernel_launch(void* const* d_in, const int* in_sizes, int n_in,
                              void* d_out, int out_size, void* d_ws, size_t ws_size,
                              hipStream_t stream)
{
    (void)in_sizes; (void)n_in; (void)out_size; (void)ws_size;
    const float* x  = (const float*)d_in[0];
    const float* Wq = (const float*)d_in[1];
    const float* bq = (const float*)d_in[2];
    const float* Wk = (const float*)d_in[3];
    const float* bk = (const float*)d_in[4];
    const float* Wv = (const float*)d_in[5];
    const float* bv = (const float*)d_in[6];

    // Workspace: 3 bf16 tensors of B*H*S*DH = 2Mi elems (4 MB) each = 12 MB.
    bf16_t* ws = (bf16_t*)d_ws;
    const size_t per = (size_t)Bb * H * S * DH;
    bf16_t* Qw = ws;
    bf16_t* Kw = ws + per;
    bf16_t* Vt = ws + 2 * per;

    dim3 gp(M_TOT / 16, 48), bp(32);
    qkv_proj_kernel<<<gp, bp, 0, stream>>>(x, Wq, bq, Wk, bk, Wv, bv, Qw, Kw, Vt);

    dim3 ga(S / 16 / 4, Bb * H), ba(128);
    flash_attn_kernel<<<ga, ba, 0, stream>>>(Qw, Kw, Vt, (float*)d_out);
}
